// Model_9337258902458
// MI455X (gfx1250) — matrix operand order repaired and checked by reading, not compiled, not run on hardware
//
#include <hip/hip_runtime.h>
#include <hip/hip_bf16.h>

// ---------------------------------------------------------------------------
// Graph-attention block for MI455X (gfx1250, wave32, WMMA + TDM).
//
// Algebraic reductions derived from the reference:
//  * Q projection eliminated (score_q constant along softmax axis -> cancels).
//  * K projection folded into a 12x768 vector bank wkp[h,:] (rank-1 use of K).
//  * score_e depends only on (b,h,j): se[b,h,j] = desc[b,j-1,h*64:].we_a.
//  * Per-row softmax/AV closed form via rank-1 diagonal correction:
//      row i>=1 : (T - e_i * v_i) / (Z - e_i),   row 0 : T / Z
// Heavy lifting left: V-GEMM and O-GEMM (4096x768x768 each) -> bf16 WMMA
// with fp32 accumulation; K-slabs moved into LDS by the Tensor Data Mover
// (tensor_load_to_lds, TENSORcnt) — this toolchain exposes the 6-arg builtin.
// ---------------------------------------------------------------------------

typedef __attribute__((ext_vector_type(16))) __bf16 v16bf;
typedef __attribute__((ext_vector_type(8)))  float  v8f;

#if defined(__has_builtin)
# if __has_builtin(__builtin_amdgcn_tensor_load_to_lds) && \
     __has_builtin(__builtin_amdgcn_s_wait_tensorcnt)
#  define USE_TDM 1
# endif
#endif
#ifndef USE_TDM
# define USE_TDM 0
#endif

#if USE_TDM
typedef __attribute__((ext_vector_type(4))) unsigned int v4u;
typedef __attribute__((ext_vector_type(8))) int          v8i;
typedef __attribute__((ext_vector_type(4))) int          v4i;
#endif

union Frag { v16bf v; unsigned u[8]; };

static __device__ __forceinline__ unsigned short f2bf(float f) {
    union { float f; unsigned u; } v; v.f = f;
    unsigned r = v.u + 0x7FFFu + ((v.u >> 16) & 1u);   // round-to-nearest-even
    return (unsigned short)(r >> 16);
}

// ---------------------------------------------------------------------------
// fp32 -> bf16 bulk convert
// ---------------------------------------------------------------------------
__global__ void cvt_kernel(const float* __restrict__ src,
                           unsigned short* __restrict__ dst, int n) {
    int i = blockIdx.x * blockDim.x + threadIdx.x;
    if (i < n) dst[i] = f2bf(src[i]);
}

// ---------------------------------------------------------------------------
// wkp[h,c] = sum_d Wk[h*64+d, c] * wk_a[d] ;  bkh[h] = bk[h*64:].wk_a
// wk_a = Wa[64..127]
// ---------------------------------------------------------------------------
__global__ void wkp_kernel(const float* __restrict__ Wk,
                           const float* __restrict__ bk,
                           const float* __restrict__ Wa,
                           float* __restrict__ wkp,
                           float* __restrict__ bkh) {
    int idx = blockIdx.x * blockDim.x + threadIdx.x;
    if (idx >= 12 * 768) return;
    int h = idx / 768, c = idx % 768;
    const float* wka = Wa + 64;
    float acc = 0.f;
    #pragma unroll 4
    for (int d = 0; d < 64; ++d)
        acc += Wk[(size_t)(h * 64 + d) * 768 + c] * wka[d];
    wkp[idx] = acc;
    if (c == 0) {
        float bb = 0.f;
        for (int d = 0; d < 64; ++d) bb += bk[h * 64 + d] * wka[d];
        bkh[h] = bb;
    }
}

// ---------------------------------------------------------------------------
// S[b,h,j] = x[b,j,:].wkp[h,:] + bkh[h] + (j>=1 ? desc[b,j-1,h*64:].we_a : -inf)
// layout: S[(b*12+h)*128 + j]
// ---------------------------------------------------------------------------
__global__ void score_kernel(const float* __restrict__ x,     // [4096][768]
                             const float* __restrict__ desc,  // [32][127][768]
                             const float* __restrict__ wkp,   // [12][768]
                             const float* __restrict__ bkh,   // [12]
                             const float* __restrict__ Wa,    // [192]
                             float* __restrict__ S) {
    int idx = blockIdx.x * blockDim.x + threadIdx.x;
    if (idx >= 32 * 128 * 12) return;
    int h   = idx % 12;
    int row = idx / 12;              // b*128 + j
    int b = row >> 7, j = row & 127;
    const float* xr = x + (size_t)row * 768;
    const float* wk = wkp + h * 768;
    float acc = bkh[h];
    #pragma unroll 4
    for (int c = 0; c < 768; ++c) acc += xr[c] * wk[c];
    if (j >= 1) {
        const float* dr = desc + ((size_t)b * 127 + (j - 1)) * 768 + h * 64;
        const float* we = Wa + 128;
        float e2 = 0.f;
        #pragma unroll 4
        for (int d = 0; d < 64; ++d) e2 += dr[d] * we[d];
        acc += e2;
    } else {
        acc = -1.0e30f;              // column j=0 is fully masked
    }
    S[(b * 12 + h) * 128 + j] = acc;
}

#if USE_TDM
// Build D# group 0: count=1, lds byte addr, 57-bit global byte addr, type=2.
static __device__ __forceinline__ v4u tdm_g0(unsigned lds_addr,
                                             unsigned long long gaddr) {
    v4u g0;
    g0.x = 1u;                                        // count=1, user mode
    g0.y = lds_addr;                                  // LDS byte address
    g0.z = (unsigned)(gaddr & 0xFFFFFFFFu);           // global_addr[31:0]
    g0.w = (unsigned)((gaddr >> 32) & 0x01FFFFFFu)    // global_addr[56:32]
         | (2u << 30);                                // type=2 ("image")
    return g0;
}
// Build D# group 1 for a 2D bf16 tile: tile_dim0 x tile_dim1 out of a tensor
// with row length tensor_dim0 (= stride) and tensor_dim1 rows.
static __device__ __forceinline__ v8i tdm_g1(unsigned tensor_d0,
                                             unsigned tensor_d1,
                                             unsigned tile_d0,
                                             unsigned tile_d1,
                                             unsigned stride_d0) {
    v8i g1;
    g1[0] = (int)(1u << 16);                      // data_size=1 -> 2 bytes
    g1[1] = (int)((tensor_d0 & 0xFFFFu) << 16);   // tensor_dim0[15:0] @ bits48
    g1[2] = (int)(((tensor_d0 >> 16) & 0xFFFFu)   // tensor_dim0[31:16]
                | ((tensor_d1 & 0xFFFFu) << 16)); // tensor_dim1[15:0] @ bits80
    g1[3] = (int)(((tensor_d1 >> 16) & 0xFFFFu)
                | ((tile_d0 & 0xFFFFu) << 16));   // tile_dim0 @ bits112
    g1[4] = (int)(tile_d1 & 0xFFFFu);             // tile_dim1 @ bits128, tile_dim2=0
    g1[5] = (int)stride_d0;                       // tensor_dim0_stride[31:0]
    g1[6] = 0;                                    // stride hi / dim1_stride lo
    g1[7] = 0;
    return g1;
}
#endif

// ---------------------------------------------------------------------------
// bf16 GEMM with fp32 accum:  C[M][768] = A[M][768](bf16) x W[768][768](bf16,
// row = output channel, i.e. B = W^T) + bias.
// Block tile 128(M) x 64(N), 256 threads = 8 wave32 waves (4x2 wave grid),
// each wave owns a 32x32 output tile = 2x2 WMMA 16x16x32 tiles.
// K-slabs moved to LDS by TDM (tensor_load_to_lds, 6-arg builtin form).
// ---------------------------------------------------------------------------
__global__ __launch_bounds__(256) void gemm_bf16_kernel(
        const unsigned short* __restrict__ A,   // bf16 [M][768]
        const unsigned short* __restrict__ W,   // bf16 [768][768]
        const float* __restrict__ bias,         // [768]
        float* __restrict__ C) {                // [M][768]
    constexpr int K = 768;
    __shared__ __align__(16) unsigned short Atile[128 * 32];
    __shared__ __align__(16) unsigned short Btile[64 * 32];

    const int t    = threadIdx.x;
    const int m0   = blockIdx.y * 128;
    const int n0   = blockIdx.x * 64;
    const int lane = t & 31;
    const int wv   = t >> 5;
    const int mb   = (wv & 3) * 32;      // wave M offset in block tile
    const int nb   = (wv >> 2) * 32;     // wave N offset in block tile
    const int am   = lane & 15;          // A: M row / B,C,D: N column (lane%16)
    const int ahi  = lane >> 4;          // lane half select

    v8f acc00 = {}, acc01 = {}, acc10 = {}, acc11 = {};

#if USE_TDM
    // Flat LDS pointers: low 32 bits are the LDS byte offset (ISA 10.2).
    const unsigned ldsA = (unsigned)(unsigned long long)(const void*)Atile;
    const unsigned ldsB = (unsigned)(unsigned long long)(const void*)Btile;
    const v4i gz4 = {0, 0, 0, 0};
    const v8i gz8 = {0, 0, 0, 0, 0, 0, 0, 0};
#endif

    for (int k0 = 0; k0 < K; k0 += 32) {
#if USE_TDM
        // ---- one TDM descriptor per tile, issued by wave 0 only ----
        if (wv == 0) {
            unsigned long long gA =
                (unsigned long long)(A + (size_t)m0 * K + k0);
            unsigned long long gB =
                (unsigned long long)(W + (size_t)n0 * K + k0);
            __builtin_amdgcn_tensor_load_to_lds(
                tdm_g0(ldsA, gA), tdm_g1(768u, 4096u, 32u, 128u, 768u),
                gz4, gz4, gz8, 0);
            __builtin_amdgcn_tensor_load_to_lds(
                tdm_g0(ldsB, gB), tdm_g1(768u, 768u, 32u, 64u, 768u),
                gz4, gz4, gz8, 0);
            __builtin_amdgcn_s_wait_tensorcnt(0);
        }
        __syncthreads();
#else
        // ---- manual staging: issue all loads, then all LDS stores ----
        {
            int r0 = t >> 2,          c0 = t & 3;           // chunk t
            int r1 = (t + 256) >> 2,  c1 = t & 3;           // chunk t+256
            uint4 ra0 = *(const uint4*)(A + (size_t)(m0 + r0) * K + k0 + c0 * 8);
            uint4 ra1 = *(const uint4*)(A + (size_t)(m0 + r1) * K + k0 + c1 * 8);
            uint4 rb  = *(const uint4*)(W + (size_t)(n0 + r0) * K + k0 + c0 * 8);
            ((uint4*)Atile)[t]       = ra0;
            ((uint4*)Atile)[t + 256] = ra1;
            ((uint4*)Btile)[t]       = rb;
        }
        if (k0 + 32 < K) {
            __builtin_prefetch(A + (size_t)(m0 + (t & 127)) * K + k0 + 32, 0, 3);
            if (t < 64)
                __builtin_prefetch(W + (size_t)(n0 + t) * K + k0 + 32, 0, 3);
        }
        __syncthreads();
#endif

        // ---- build WMMA fragments per ISA 7.12.2 layouts ----
        Frag a0, a1, b0, b1;
        #pragma unroll
        for (int v = 0; v < 8; ++v) {
            // A 16x32 bf16: lane half picks K block of 8; pairs per VGPR
            int ka = ((v & 4) ? 16 : 0) + ahi * 8 + ((v & 3) << 1);
            a0.u[v] = *(const unsigned*)&Atile[(mb +      am) * 32 + ka];
            a1.u[v] = *(const unsigned*)&Atile[(mb + 16 + am) * 32 + ka];
            // B 32x16 bf16: same K order as A (lane half picks K block of 8)
            int kb = ((v & 4) ? 16 : 0) + ahi * 8 + ((v & 3) << 1);
            b0.u[v] = *(const unsigned*)&Btile[(nb +      am) * 32 + kb];
            b1.u[v] = *(const unsigned*)&Btile[(nb + 16 + am) * 32 + kb];
        }

        acc00 = __builtin_amdgcn_wmma_f32_16x16x32_bf16(false, a0.v, false, b0.v,
                                                        (short)0, acc00, false, false);
        acc01 = __builtin_amdgcn_wmma_f32_16x16x32_bf16(false, a0.v, false, b1.v,
                                                        (short)0, acc01, false, false);
        acc10 = __builtin_amdgcn_wmma_f32_16x16x32_bf16(false, a1.v, false, b0.v,
                                                        (short)0, acc10, false, false);
        acc11 = __builtin_amdgcn_wmma_f32_16x16x32_bf16(false, a1.v, false, b1.v,
                                                        (short)0, acc11, false, false);
        __syncthreads();
    }

    // ---- epilogue: C/D layout -> lanes 0-15: M=r, lanes 16-31: M=8+r; N=lane%16
    const int mhi = ahi * 8;
    const float bias0 = bias[n0 + nb + am];
    const float bias1 = bias[n0 + nb + 16 + am];
    #pragma unroll
    for (int r = 0; r < 8; ++r) {
        int mg0 = m0 + mb +      mhi + r;
        int mg1 = m0 + mb + 16 + mhi + r;
        int ng0 = n0 + nb + am;
        int ng1 = n0 + nb + 16 + am;
        C[(size_t)mg0 * 768 + ng0] = acc00[r] + bias0;
        C[(size_t)mg0 * 768 + ng1] = acc01[r] + bias1;
        C[(size_t)mg1 * 768 + ng0] = acc10[r] + bias0;
        C[(size_t)mg1 * 768 + ng1] = acc11[r] + bias1;
    }
}

// ---------------------------------------------------------------------------
// Per-(b,h) attention: softmax over the shared score row + rank-1 diagonal
// correction, writes bf16 (feeds the O-projection WMMA GEMM).
// ---------------------------------------------------------------------------
__global__ __launch_bounds__(256) void attn_kernel(
        const float* __restrict__ S,          // [(b*12+h)][128]
        const float* __restrict__ V,          // [4096][768] fp32
        unsigned short* __restrict__ Ob) {    // [4096][768] bf16
    __shared__ float s[128], e[128], red[64], Tp[4][64], T[64];
    const int bh = blockIdx.x;                // 0..383
    const int b = bh / 12, h = bh % 12;
    const int t = threadIdx.x;

    if (t < 128) s[t] = S[bh * 128 + t];
    __syncthreads();

    // max over j=1..127
    if (t < 64) {
        float a = (t >= 1) ? s[t] : -3.0e38f;
        red[t] = fmaxf(a, s[t + 64]);
    }
    __syncthreads();
    for (int w = 32; w >= 1; w >>= 1) {
        if (t < w) red[t] = fmaxf(red[t], red[t + w]);
        __syncthreads();
    }
    const float m = red[0];
    __syncthreads();

    // e_j and Z
    if (t < 128) e[t] = (t >= 1) ? __expf(s[t] - m) : 0.f;
    __syncthreads();
    if (t < 64) red[t] = e[t] + e[t + 64];
    __syncthreads();
    for (int w = 32; w >= 1; w >>= 1) {
        if (t < w) red[t] += red[t + w];
        __syncthreads();
    }
    const float Z = red[0];

    // T[d] = sum_j e_j * v[j,d]
    const int d = t & 63, jg = t >> 6;
    const float* vb = V + ((size_t)b * 128) * 768 + h * 64 + d;
    float part = 0.f;
    for (int j = jg; j < 128; j += 4) part += e[j] * vb[(size_t)j * 768];
    Tp[jg][d] = part;
    __syncthreads();
    if (t < 64) T[t] = Tp[0][t] + Tp[1][t] + Tp[2][t] + Tp[3][t];
    __syncthreads();

    // out rows: (T - e_i v_i)/(Z - e_i), row 0: T/Z
    for (int i = jg; i < 128; i += 4) {
        float num, den;
        if (i == 0) { num = T[d];                          den = Z; }
        else        { float ei = e[i];
                      num = T[d] - ei * vb[(size_t)i * 768]; den = Z - ei; }
        Ob[((size_t)b * 128 + i) * 768 + h * 64 + d] = f2bf(num / den);
    }
}

// ---------------------------------------------------------------------------
// Host-side launch
// ---------------------------------------------------------------------------
extern "C" void kernel_launch(void* const* d_in, const int* in_sizes, int n_in,
                              void* d_out, int out_size, void* d_ws, size_t ws_size,
                              hipStream_t stream) {
    const float* desc = (const float*)d_in[0];   // (32,127,768)
    const float* x    = (const float*)d_in[1];   // (32,128,768)
    // d_in[2]=Wq, d_in[3]=bq : eliminated (softmax shift invariance)
    const float* Wk = (const float*)d_in[4];
    const float* bk = (const float*)d_in[5];
    const float* Wv = (const float*)d_in[6];
    const float* bv = (const float*)d_in[7];
    const float* Wo = (const float*)d_in[8];
    const float* bo = (const float*)d_in[9];
    const float* Wa = (const float*)d_in[10];
    // d_in[11]=ba : eliminated (constant along softmax axis)
    float* out = (float*)d_out;

    char* w = (char*)d_ws;
    unsigned short* xb  = (unsigned short*)w; w += 4096ull * 768 * 2;  // x bf16
    unsigned short* Wvb = (unsigned short*)w; w += 768ull * 768 * 2;
    unsigned short* Wob = (unsigned short*)w; w += 768ull * 768 * 2;
    unsigned short* Ab  = (unsigned short*)w; w += 4096ull * 768 * 2;  // attn-out bf16
    float* Vf  = (float*)w; w += 4096ull * 768 * 4;
    float* wkp = (float*)w; w += 12ull * 768 * 4;
    float* bkh = (float*)w; w += 256;
    float* S   = (float*)w; w += 384ull * 128 * 4;

    cvt_kernel<<<(4096 * 768 + 255) / 256, 256, 0, stream>>>(x, xb, 4096 * 768);
    cvt_kernel<<<(768 * 768 + 255) / 256, 256, 0, stream>>>(Wv, Wvb, 768 * 768);
    cvt_kernel<<<(768 * 768 + 255) / 256, 256, 0, stream>>>(Wo, Wob, 768 * 768);

    wkp_kernel<<<(12 * 768 + 255) / 256, 256, 0, stream>>>(Wk, bk, Wa, wkp, bkh);
    score_kernel<<<(32 * 128 * 12 + 255) / 256, 256, 0, stream>>>(x, desc, wkp, bkh, Wa, S);

    dim3 gemm_grid(12, 32);   // N/64 x M/128
    gemm_bf16_kernel<<<gemm_grid, 256, 0, stream>>>(xb, Wvb, bv, Vf);

    attn_kernel<<<384, 256, 0, stream>>>(S, Vf, Ab);

    gemm_bf16_kernel<<<gemm_grid, 256, 0, stream>>>(Ab, Wob, bo, out);
}
